// VSSBlock_79663053406389
// MI455X (gfx1250) — matrix operand order repaired and checked by reading, not compiled, not run on hardware
//
#include <hip/hip_runtime.h>
#include <hip/hip_bf16.h>

// ---------------------------------------------------------------------------
// VSS block (VMamba) forward for MI455X / gfx1250, wave32, WMMA f16 tensor ops
// Shapes: B=4, H=W=64, C=128, Di=256, Kdir=4, Nst=16, R=8, L=4096, M=16384
// Selective scan parallelized over 16 L-segments (linear diagonal recurrence).
// ---------------------------------------------------------------------------

typedef __attribute__((ext_vector_type(16))) _Float16 v16h;
typedef __attribute__((ext_vector_type(8)))  float    v8f;

#define B_    4
#define HW_   64
#define C_    128
#define DI_   256
#define KD_   4
#define NST_  16
#define R_    8
#define L_    4096
#define M_    16384        // B_*L_
#define CM_   512
#define NPAD_ 48           // R + 2N = 40 padded to 48 (3 tiles of 16)
#define SEGS_ 16
#define SEGL_ 256          // L_/SEGS_

// ---------------- fragment helpers (contiguous-k A/B layouts, wave32) ------------

__device__ __forceinline__ v16h mk_v16h(uint4 q0, uint4 q1) {
  union { v16h h; uint4 q[2]; } u;
  u.q[0] = q0; u.q[1] = q1;
  return u.h;
}

// A 16x32 f16 frag: lane half=lid>>4, row m=lid&15.
// VGPR v pairs: K = 2*v + 16*half -> two contiguous 16B loads.
__device__ __forceinline__ v16h load_fragA(const _Float16* __restrict__ arow,
                                           int kk, unsigned half) {
  uint4 q0 = *(const uint4*)(arow + kk + 16 * half);
  uint4 q1 = *(const uint4*)(arow + kk + 16 * half + 8);
  return mk_v16h(q0, q1);
}

// B 32x16 f16 frag: lane col n=lid&15, half=lid>>4.
// VGPR v pairs: K = 2*v + 16*half -> two contiguous 16B loads.
// Bw stored as [n][k] row-major (weight[n][k], k contiguous).
__device__ __forceinline__ v16h load_fragB(const _Float16* __restrict__ brow,
                                           int kk) {
  uint4 q0 = *(const uint4*)(brow + kk);
  uint4 q1 = *(const uint4*)(brow + kk + 8);
  return mk_v16h(q0, q1);
}

__device__ __forceinline__ float gelu_tanh(float x) {
  float x3 = x * x * x;
  return 0.5f * x * (1.0f + tanhf(0.7978845608028654f * (x + 0.044715f * x3)));
}
__device__ __forceinline__ float silu(float x) { return x / (1.0f + expf(-x)); }
__device__ __forceinline__ float softplus(float x) {
  return (x > 20.0f) ? x : log1pf(expf(x));
}
// scan position of step i for direction k (H=W=64 transpose is an involution)
__device__ __forceinline__ int scan_pos(int k, int i) {
  int j = (k & 2) ? (L_ - 1 - i) : i;
  return (k & 1) ? (((j & 63) << 6) | (j >> 6)) : j;
}

// ---------------- weight casts ---------------------------------------------

__global__ void cast_f16_k(const float* __restrict__ src,
                           _Float16* __restrict__ dst, int n) {
  for (int i = blockIdx.x * blockDim.x + threadIdx.x; i < n;
       i += gridDim.x * blockDim.x)
    dst[i] = (_Float16)src[i];
}

// x_proj_w (K,40,Di) -> padded f16 (K,48,Di)
__global__ void cast_xproj_k(const float* __restrict__ src,
                             _Float16* __restrict__ dst) {
  int i = blockIdx.x * blockDim.x + threadIdx.x;
  if (i >= KD_ * NPAD_ * DI_) return;
  int d = i & (DI_ - 1);
  int c = (i >> 8) % NPAD_;
  int k = i / (NPAD_ * DI_);
  dst[i] = (_Float16)((c < 40) ? src[((size_t)(k * 40 + c)) * DI_ + d] : 0.0f);
}

// ---------------- LayerNorm over 128, one wave per row, f16 out ------------

__global__ __launch_bounds__(256)
void ln128_k(const float* __restrict__ X, const float* __restrict__ w,
             const float* __restrict__ b, _Float16* __restrict__ out, int rows) {
  int row = blockIdx.x * 8 + (threadIdx.x >> 5);
  if (row >= rows) return;
  int lid = threadIdx.x & 31;
  const float* xr = X + (size_t)row * C_;
  float v[4], s = 0.f, s2 = 0.f;
#pragma unroll
  for (int j = 0; j < 4; ++j) {
    v[j] = xr[lid + 32 * j];
    s += v[j]; s2 += v[j] * v[j];
  }
#pragma unroll
  for (int m = 16; m; m >>= 1) { s += __shfl_xor(s, m); s2 += __shfl_xor(s2, m); }
  float mu = s * (1.0f / C_);
  float var = s2 * (1.0f / C_) - mu * mu;
  float r = rsqrtf(var + 1e-5f);
#pragma unroll
  for (int j = 0; j < 4; ++j) {
    int c = lid + 32 * j;
    out[(size_t)row * C_ + c] = (_Float16)((v[j] - mu) * r * w[c] + b[c]);
  }
}

// ---------------- generic WMMA GEMM: C = A(MxK) * Bw(NxK)^T ----------------
// Each wave: one A fragment x NT B fragments -> NT accumulators (A reuse,
// NT independent WMMA chains per k-step to cover the WMMA->WMMA hazard).
// EPI 0: out_f32 = acc                         (in_proj -> xz)
// EPI 1: out_f32 = acc + aux0[m*N+n]           (out_proj + residual x)
// EPI 2: out_f16 = gelu(acc + aux0[n])         (fc1)
// EPI 3: out_f32 = acc + aux0[n] + aux1[m*N+n] (fc2 + bias + residual)

template <int EPI, int NT>
__global__ __launch_bounds__(256)
void gemm_wmma_k(const _Float16* __restrict__ A, const _Float16* __restrict__ Bw,
                 float* __restrict__ outF, _Float16* __restrict__ outH,
                 const float* __restrict__ aux0, const float* __restrict__ aux1,
                 int M, int N, int Kd) {
  const int wave = blockIdx.x * (blockDim.x >> 5) + (threadIdx.x >> 5);
  const int nGroups = (N >> 4) / NT;
  const int total = (M >> 4) * nGroups;
  if (wave >= total) return;                 // wave-uniform: EXEC stays full
  const int mt = wave / nGroups, ng = wave % nGroups;
  const unsigned lid = threadIdx.x & 31u;
  const unsigned half = lid >> 4, mn = lid & 15u;
  const _Float16* arow = A + (size_t)(mt * 16 + mn) * Kd;
  const _Float16* brow[NT];
#pragma unroll
  for (int j = 0; j < NT; ++j)
    brow[j] = Bw + (size_t)((ng * NT + j) * 16 + mn) * Kd + 16 * half;
  v8f acc[NT];
#pragma unroll
  for (int j = 0; j < NT; ++j) acc[j] = (v8f){};
  for (int kk = 0; kk < Kd; kk += 32) {
    __builtin_prefetch(arow + kk + 128, 0, 1);   // global_prefetch_b8
    v16h fa = load_fragA(arow, kk, half);
#pragma unroll
    for (int j = 0; j < NT; ++j) {
      v16h fb = load_fragB(brow[j], kk);
      acc[j] = __builtin_amdgcn_wmma_f32_16x16x32_f16(false, fa, false, fb,
                                                      (short)0, acc[j],
                                                      false, false);
    }
  }
#pragma unroll
  for (int j = 0; j < NT; ++j) {
    const int col = (ng * NT + j) * 16 + mn;
#pragma unroll
    for (int i = 0; i < 8; ++i) {
      int row = mt * 16 + i + 8 * (int)half;   // C/D layout (ISA 7.12.2)
      size_t o = (size_t)row * N + col;
      float v = acc[j][i];
      if constexpr (EPI == 0) outF[o] = v;
      else if constexpr (EPI == 1) outF[o] = v + aux0[o];
      else if constexpr (EPI == 2) outH[o] = (_Float16)gelu_tanh(v + aux0[col]);
      else outF[o] = v + aux0[col] + aux1[o];
    }
  }
}

// ---------------- x_proj GEMM with per-direction A-row remap ---------------
// One wave covers the whole 48-wide output strip (3 B frags per A frag).

__global__ __launch_bounds__(256)
void gemm_xproj_k(const _Float16* __restrict__ xcH,
                  const _Float16* __restrict__ Wxp,
                  float* __restrict__ xdbl) {
  const int wave = blockIdx.x * 8 + (threadIdx.x >> 5);
  const int total = (B_ * KD_) * (L_ / 16);         // 4096
  if (wave >= total) return;
  int mt = wave & 255;
  int bk = wave >> 8;
  int b = bk >> 2, k = bk & 3;
  const unsigned lid = threadIdx.x & 31u;
  const unsigned half = lid >> 4, mn = lid & 15u;
  int i = mt * 16 + mn;
  int p = scan_pos(k, i);
  const _Float16* arow = xcH + (size_t)(b * L_ + p) * DI_;
  const _Float16* brow[3];
#pragma unroll
  for (int j = 0; j < 3; ++j)
    brow[j] = Wxp + (size_t)(k * NPAD_ + j * 16 + mn) * DI_ + 16 * half;
  v8f acc[3];
#pragma unroll
  for (int j = 0; j < 3; ++j) acc[j] = (v8f){};
  for (int kk = 0; kk < DI_; kk += 32) {
    v16h fa = load_fragA(arow, kk, half);
#pragma unroll
    for (int j = 0; j < 3; ++j) {
      v16h fb = load_fragB(brow[j], kk);
      acc[j] = __builtin_amdgcn_wmma_f32_16x16x32_f16(false, fa, false, fb,
                                                      (short)0, acc[j],
                                                      false, false);
    }
  }
#pragma unroll
  for (int j = 0; j < 3; ++j) {
    const int col = j * 16 + mn;
#pragma unroll
    for (int ii = 0; ii < 8; ++ii) {
      int row = mt * 16 + ii + 8 * (int)half;
      xdbl[((size_t)bk * L_ + row) * NPAD_ + col] = acc[j][ii];
    }
  }
}

// ---------------- depthwise 3x3 conv + SiLU --------------------------------
// in: xz rows (b*L+l)*512, cols 0..255 = xx.  out: xc (B,L,Di) f32 + f16

__global__ __launch_bounds__(256)
void conv_silu_k(const float* __restrict__ xz, const float* __restrict__ cw,
                 const float* __restrict__ cb, float* __restrict__ xc,
                 _Float16* __restrict__ xcH) {
  int blk = blockIdx.x;                 // b*L + l
  int l = blk & (L_ - 1);
  int h = l >> 6, w = l & 63;
  int d = threadIdx.x;
  int base = blk - l;                   // b*L
  float acc = cb[d];
#pragma unroll
  for (int i = 0; i < 3; ++i) {
    int hh = h + i - 1;
    if ((unsigned)hh >= 64u) continue;
#pragma unroll
    for (int j = 0; j < 3; ++j) {
      int ww = w + j - 1;
      if ((unsigned)ww >= 64u) continue;
      acc = fmaf(cw[d * 9 + i * 3 + j],
                 xz[((size_t)(base + hh * 64 + ww)) * CM_ + d], acc);
    }
  }
  float s = silu(acc);
  xc[(size_t)blk * DI_ + d] = s;
  xcH[(size_t)blk * DI_ + d] = (_Float16)s;
}

// ---------------- dt projection (K=8) + softplus ---------------------------

__global__ __launch_bounds__(256)
void dtproj_k(const float* __restrict__ xdbl, const float* __restrict__ Wdt,
              const float* __restrict__ bdt, float* __restrict__ dtb) {
  int blk = blockIdx.x;                 // bk*L + l
  int k = (blk >> 12) & 3;
  __shared__ float sdt[R_];
  if (threadIdx.x < R_) sdt[threadIdx.x] = xdbl[(size_t)blk * NPAD_ + threadIdx.x];
  __syncthreads();
  int d = threadIdx.x;
  const float* wr = Wdt + (size_t)(k * DI_ + d) * R_;
  float a = bdt[k * DI_ + d];
#pragma unroll
  for (int r = 0; r < R_; ++r) a = fmaf(sdt[r], wr[r], a);
  dtb[(size_t)blk * DI_ + d] = softplus(a);
}

// ---------------- selective scan, segmented (linear diagonal SSM) ----------
// h_l = exp(a*dt_l) h_{l-1} + dt_l u_l B_l ;  y_l = <h_l, C_l> + D u_l
// Split L into SEGS_ segments: pass1 local scan (h_in=0) + segment summary,
// pass2 16-step prefix over segments, pass3 per-step correction
//   y_l += sum_n h_in[n] * exp(a_n * cumdt_l) * C_l[n].

// pass 1: 1024 blocks (bk,chunk,seg) x 64 threads
__global__ __launch_bounds__(64)
void scan_seg_k(const float* __restrict__ xdbl, const float* __restrict__ dtb,
                const float* __restrict__ xc, const float* __restrict__ Alogs,
                const float* __restrict__ Ds, float* __restrict__ ysb,
                float* __restrict__ hseg, float* __restrict__ dtsum_seg) {
  int seg = blockIdx.x & (SEGS_ - 1);
  int chunk = (blockIdx.x >> 4) & 3;
  int bk = blockIdx.x >> 6;
  int b = bk >> 2, k = bk & 3;
  int d = chunk * 64 + threadIdx.x;
  float a[NST_], h[NST_];
#pragma unroll
  for (int n = 0; n < NST_; ++n) {
    a[n] = -__expf(Alogs[(size_t)(k * DI_ + d) * NST_ + n]);
    h[n] = 0.0f;
  }
  float Dv = Ds[k * DI_ + d];
  float dtsum = 0.0f;
  __shared__ float BC[32 * 32];         // [step][c], 0..15=B, 16..31=C
  const float* xdb = xdbl + (size_t)bk * L_ * NPAD_;
  const int lbeg = seg * SEGL_;
  for (int l0 = lbeg; l0 < lbeg + SEGL_; l0 += 32) {
    __syncthreads();
    for (int t = threadIdx.x; t < 1024; t += 64) {
      int st = t >> 5, c = t & 31;
      BC[st * 32 + c] = xdb[(size_t)(l0 + st) * NPAD_ + R_ + c];
    }
    __syncthreads();
    for (int i = 0; i < 32; ++i) {
      int l = l0 + i;
      int p = scan_pos(k, l);
      float dt = dtb[((size_t)bk * L_ + l) * DI_ + d];
      float u  = xc[((size_t)(b * L_ + p)) * DI_ + d];
      dtsum += dt;
      float du = dt * u;
      float y = Dv * u;
      const float* Bs = &BC[i * 32];
      const float* Cs = Bs + 16;
#pragma unroll
      for (int n = 0; n < NST_; ++n) {
        h[n] = fmaf(h[n], __expf(dt * a[n]), du * Bs[n]);
        y = fmaf(h[n], Cs[n], y);
      }
      ysb[((size_t)bk * L_ + l) * DI_ + d] = y;
    }
  }
  size_t hb = (((size_t)bk * DI_ + d) * SEGS_ + seg) * NST_;
#pragma unroll
  for (int n = 0; n < NST_; ++n) hseg[hb + n] = h[n];
  dtsum_seg[((size_t)bk * DI_ + d) * SEGS_ + seg] = dtsum;
}

// pass 2: prefix over 16 segments; 16 blocks x 256 threads (one per (bk,d))
__global__ __launch_bounds__(256)
void seg_prefix_k(const float* __restrict__ Alogs,
                  const float* __restrict__ hseg,
                  const float* __restrict__ dtsum_seg,
                  float* __restrict__ hin) {
  int bk = blockIdx.x, k = bk & 3, d = threadIdx.x;
  float a[NST_], h[NST_];
#pragma unroll
  for (int n = 0; n < NST_; ++n) {
    a[n] = -__expf(Alogs[(size_t)(k * DI_ + d) * NST_ + n]);
    h[n] = 0.0f;
  }
  size_t base = (size_t)bk * DI_ + d;
  for (int seg = 0; seg < SEGS_; ++seg) {
    size_t hb = (base * SEGS_ + seg) * NST_;
#pragma unroll
    for (int n = 0; n < NST_; ++n) hin[hb + n] = h[n];
    float ds = dtsum_seg[base * SEGS_ + seg];
#pragma unroll
    for (int n = 0; n < NST_; ++n)
      h[n] = fmaf(h[n], __expf(a[n] * ds), hseg[hb + n]);
  }
}

// pass 3: correction for segments 1..15; same grid shape as pass 1
__global__ __launch_bounds__(64)
void scan_fix_k(const float* __restrict__ xdbl, const float* __restrict__ dtb,
                const float* __restrict__ Alogs, const float* __restrict__ hin,
                float* __restrict__ ysb) {
  int seg = blockIdx.x & (SEGS_ - 1);
  if (seg == 0) return;                  // block-uniform
  int chunk = (blockIdx.x >> 4) & 3;
  int bk = blockIdx.x >> 6;
  int k = bk & 3;
  int d = chunk * 64 + threadIdx.x;
  float a[NST_], hi[NST_];
  size_t hb = (((size_t)bk * DI_ + d) * SEGS_ + seg) * NST_;
#pragma unroll
  for (int n = 0; n < NST_; ++n) {
    a[n] = -__expf(Alogs[(size_t)(k * DI_ + d) * NST_ + n]);
    hi[n] = hin[hb + n];
  }
  __shared__ float Csh[32 * 16];
  const float* xdb = xdbl + (size_t)bk * L_ * NPAD_;
  float cumdt = 0.0f;
  const int lbeg = seg * SEGL_;
  for (int l0 = lbeg; l0 < lbeg + SEGL_; l0 += 32) {
    __syncthreads();
    for (int t = threadIdx.x; t < 512; t += 64) {
      int st = t >> 4, c = t & 15;
      Csh[st * 16 + c] = xdb[(size_t)(l0 + st) * NPAD_ + R_ + NST_ + c];
    }
    __syncthreads();
    for (int i = 0; i < 32; ++i) {
      int l = l0 + i;
      float dt = dtb[((size_t)bk * L_ + l) * DI_ + d];
      cumdt += dt;
      float corr = 0.0f;
      const float* Cs = &Csh[i * 16];
#pragma unroll
      for (int n = 0; n < NST_; ++n)
        corr = fmaf(hi[n] * __expf(a[n] * cumdt), Cs[n], corr);
      size_t o = ((size_t)bk * L_ + l) * DI_ + d;
      ysb[o] += corr;
    }
  }
}

// ---------------- cross-merge + out_norm(Di=256) + SiLU gate ---------------

__global__ __launch_bounds__(256)
void merge_gate_k(const float* __restrict__ ys, const float* __restrict__ xz,
                  const float* __restrict__ onw, const float* __restrict__ onb,
                  _Float16* __restrict__ ygH) {
  int blk = blockIdx.x;                 // b*L + l
  int b = blk >> 12, l = blk & (L_ - 1);
  int d = threadIdx.x;
  int lt = ((l & 63) << 6) | (l >> 6);
  size_t s0 = ((size_t)((b * 4 + 0) * L_ + l)) * DI_ + d;
  size_t s1 = ((size_t)((b * 4 + 1) * L_ + lt)) * DI_ + d;
  size_t s2 = ((size_t)((b * 4 + 2) * L_ + (L_ - 1 - l))) * DI_ + d;
  size_t s3 = ((size_t)((b * 4 + 3) * L_ + (L_ - 1 - lt))) * DI_ + d;
  float y = ys[s0] + ys[s1] + ys[s2] + ys[s3];

  __shared__ float red[16];
  float s = y, s2v = y * y;
#pragma unroll
  for (int m = 16; m; m >>= 1) { s += __shfl_xor(s, m); s2v += __shfl_xor(s2v, m); }
  int wv = threadIdx.x >> 5;
  if ((threadIdx.x & 31) == 0) { red[wv] = s; red[8 + wv] = s2v; }
  __syncthreads();
  if (threadIdx.x == 0) {
    float S = 0.f, S2 = 0.f;
#pragma unroll
    for (int wvi = 0; wvi < 8; ++wvi) { S += red[wvi]; S2 += red[8 + wvi]; }
    float mu = S * (1.0f / DI_);
    float var = S2 * (1.0f / DI_) - mu * mu;
    red[0] = mu; red[1] = rsqrtf(var + 1e-5f);
  }
  __syncthreads();
  float mu = red[0], r = red[1];
  float ln = (y - mu) * r * onw[d] + onb[d];
  float z = xz[(size_t)blk * CM_ + DI_ + d];
  ygH[(size_t)blk * DI_ + d] = (_Float16)(ln * silu(z));
}

// ---------------------------------------------------------------------------

extern "C" void kernel_launch(void* const* d_in, const int* in_sizes, int n_in,
                              void* d_out, int out_size, void* d_ws, size_t ws_size,
                              hipStream_t stream) {
  const float* x        = (const float*)d_in[0];
  const float* norm1_w  = (const float*)d_in[1];
  const float* norm1_b  = (const float*)d_in[2];
  const float* in_proj  = (const float*)d_in[3];
  const float* conv_w   = (const float*)d_in[4];
  const float* conv_b   = (const float*)d_in[5];
  const float* x_proj_w = (const float*)d_in[6];
  const float* dt_w     = (const float*)d_in[7];
  const float* dt_b     = (const float*)d_in[8];
  const float* A_logs   = (const float*)d_in[9];
  const float* Ds       = (const float*)d_in[10];
  const float* onw      = (const float*)d_in[11];
  const float* onb      = (const float*)d_in[12];
  const float* out_proj = (const float*)d_in[13];
  const float* norm2_w  = (const float*)d_in[14];
  const float* norm2_b  = (const float*)d_in[15];
  const float* fc1_w    = (const float*)d_in[16];
  const float* fc1_b    = (const float*)d_in[17];
  const float* fc2_w    = (const float*)d_in[18];
  const float* fc2_b    = (const float*)d_in[19];
  float* out = (float*)d_out;

  char* ws = (char*)d_ws;
  size_t off = 0;
  auto alloc = [&](size_t bytes) {
    char* p = ws + off;
    off = (off + bytes + 255) & ~(size_t)255;
    return p;
  };
  _Float16* w16_in  = (_Float16*)alloc((size_t)512 * 128 * 2);
  _Float16* w16_xp  = (_Float16*)alloc((size_t)KD_ * NPAD_ * DI_ * 2);
  _Float16* w16_out = (_Float16*)alloc((size_t)128 * 256 * 2);
  _Float16* w16_fc1 = (_Float16*)alloc((size_t)512 * 128 * 2);
  _Float16* w16_fc2 = (_Float16*)alloc((size_t)128 * 512 * 2);
  _Float16* xn_h    = (_Float16*)alloc((size_t)M_ * C_ * 2);
  float*    xz      = (float*)   alloc((size_t)M_ * CM_ * 4);
  float*    xc      = (float*)   alloc((size_t)M_ * DI_ * 4);
  _Float16* xc_h    = (_Float16*)alloc((size_t)M_ * DI_ * 2);
  float*    xdbl    = (float*)   alloc((size_t)B_ * KD_ * L_ * NPAD_ * 4);
  float*    dtb     = (float*)   alloc((size_t)B_ * KD_ * L_ * DI_ * 4);
  float*    ysb     = (float*)   alloc((size_t)B_ * KD_ * L_ * DI_ * 4);
  _Float16* yg_h    = (_Float16*)alloc((size_t)M_ * DI_ * 2);
  float*    xres    = (float*)   alloc((size_t)M_ * C_ * 4);
  _Float16* xm_h    = (_Float16*)alloc((size_t)M_ * C_ * 2);
  _Float16* hbuf    = (_Float16*)alloc((size_t)M_ * CM_ * 2);
  float*    hseg    = (float*)   alloc((size_t)16 * DI_ * SEGS_ * NST_ * 4);
  float*    dtsumb  = (float*)   alloc((size_t)16 * DI_ * SEGS_ * 4);
  float*    hinb    = (float*)   alloc((size_t)16 * DI_ * SEGS_ * NST_ * 4);
  (void)ws_size; (void)n_in; (void)in_sizes; (void)out_size;

  // --- weight casts to f16 ---
  cast_f16_k<<<256, 256, 0, stream>>>(in_proj,  w16_in,  512 * 128);
  cast_f16_k<<<128, 256, 0, stream>>>(out_proj, w16_out, 128 * 256);
  cast_f16_k<<<256, 256, 0, stream>>>(fc1_w,    w16_fc1, 512 * 128);
  cast_f16_k<<<256, 256, 0, stream>>>(fc2_w,    w16_fc2, 128 * 512);
  cast_xproj_k<<<(KD_ * NPAD_ * DI_ + 255) / 256, 256, 0, stream>>>(x_proj_w, w16_xp);

  // --- SS2D branch ---
  ln128_k<<<M_ / 8, 256, 0, stream>>>(x, norm1_w, norm1_b, xn_h, M_);

  // in_proj: (M,128)x(512,128)^T -> xz (M,512)   [NT=4 -> 8 n-groups]
  gemm_wmma_k<0, 4><<<(M_ / 16) * (CM_ / 64) / 8, 256, 0, stream>>>(
      xn_h, w16_in, xz, nullptr, nullptr, nullptr, M_, CM_, C_);

  conv_silu_k<<<M_, 256, 0, stream>>>(xz, conv_w, conv_b, xc, xc_h);

  gemm_xproj_k<<<(B_ * KD_ * (L_ / 16)) / 8, 256, 0, stream>>>(xc_h, w16_xp, xdbl);

  dtproj_k<<<B_ * KD_ * L_, 256, 0, stream>>>(xdbl, dt_w, dt_b, dtb);

  // segmented selective scan: local scans -> segment prefix -> correction
  scan_seg_k<<<16 * 4 * SEGS_, 64, 0, stream>>>(xdbl, dtb, xc, A_logs, Ds,
                                                ysb, hseg, dtsumb);
  seg_prefix_k<<<16, 256, 0, stream>>>(A_logs, hseg, dtsumb, hinb);
  scan_fix_k<<<16 * 4 * SEGS_, 64, 0, stream>>>(xdbl, dtb, A_logs, hinb, ysb);

  merge_gate_k<<<M_, 256, 0, stream>>>(ysb, xz, onw, onb, yg_h);

  // out_proj + residual x: (M,256)x(128,256)^T + x -> xres (M,128)
  gemm_wmma_k<1, 4><<<(M_ / 16) * (C_ / 64) / 8, 256, 0, stream>>>(
      yg_h, w16_out, xres, nullptr, x, nullptr, M_, C_, DI_);

  // --- MLP branch ---
  ln128_k<<<M_ / 8, 256, 0, stream>>>(xres, norm2_w, norm2_b, xm_h, M_);

  // fc1 + bias + gelu -> hbuf f16 (M,512)
  gemm_wmma_k<2, 4><<<(M_ / 16) * (CM_ / 64) / 8, 256, 0, stream>>>(
      xm_h, w16_fc1, nullptr, hbuf, fc1_b, nullptr, M_, CM_, C_);

  // fc2 + bias + residual -> out (M,128)
  gemm_wmma_k<3, 4><<<(M_ / 16) * (C_ / 64) / 8, 256, 0, stream>>>(
      hbuf, w16_fc2, out, nullptr, fc2_b, xres, M_, C_, CM_);
}
